// Attention_11020886081921
// MI455X (gfx1250) — compile-verified, matrix operand order checked by reading, not run on hardware
//
#include <hip/hip_runtime.h>
#include <hip/hip_bf16.h>

typedef __attribute__((ext_vector_type(16))) _Float16 v16h;
typedef __attribute__((ext_vector_type(8)))  _Float16 v8h;
typedef __attribute__((ext_vector_type(2)))  _Float16 v2h;
typedef __attribute__((ext_vector_type(4)))  _Float16 v4h;
typedef __attribute__((ext_vector_type(8)))  float    v8f;
typedef int v4i_g __attribute__((vector_size(16)));   // matches async builtin param
typedef _Float16 half_t;

#define WMMA_F16(A, B, C) \
  __builtin_amdgcn_wmma_f32_16x16x32_f16(false, (A), false, (B), (short)0, (C), false, false)

// gfx1250 async global->LDS copy path (guarded; plain LDS store fallback)
#if defined(__has_builtin)
#if __has_builtin(__builtin_amdgcn_global_load_async_to_lds_b128) && \
    __has_builtin(__builtin_amdgcn_s_wait_asynccnt)
#define USE_ASYNC_LDS 1
#endif
#endif
#ifndef USE_ASYNC_LDS
#define USE_ASYNC_LDS 0
#endif
#define AS1 __attribute__((address_space(1)))
#define AS3 __attribute__((address_space(3)))

// Problem constants
#define Bx   64
#define Nx   320
#define Cx   768
#define Hx   12
#define Dx   64
#define M_TOK (Bx * Nx)          // 20480
#define QKV_N (3 * Cx)           // 2304
// LDS row pitch in halfs: 32 data + 8 pad -> 80B rows, bank stride 20,
// conflict-free across 16-lane groups, still 16B-aligned for b128 ops.
#define LDSP 40
// scale = 1/sqrt(64) * log2(e): softmax runs in exp2 domain
#define Q_SCALE 0.18033688011112042f

// Combine two 16B-aligned LDS chunks into one WMMA fragment
union HV { v16h v; v8h h[2]; };
static __device__ inline v16h lds_frag(const half_t* p) {
  HV u;
  u.h[0] = *(const v8h*)(p);
  u.h[1] = *(const v8h*)(p + 8);
  return u.v;
}

// ---------------------------------------------------------------------------
// f32 -> f16 conversion, vectorized 4-wide (all sizes divisible by 1024)
// ---------------------------------------------------------------------------
__global__ void cvt_f32_to_f16(const float* __restrict__ in,
                               half_t* __restrict__ out, int n4) {
  int i = blockIdx.x * blockDim.x + threadIdx.x;
  int stride = gridDim.x * blockDim.x;
  for (; i < n4; i += stride) {
    const float4 f = ((const float4*)in)[i];
    v4h h; h.x = (half_t)f.x; h.y = (half_t)f.y; h.z = (half_t)f.z; h.w = (half_t)f.w;
    *(v4h*)(out + (size_t)i * 4) = h;
  }
}

// ---------------------------------------------------------------------------
// Shared GEMM core: 8 waves/block share one 32-col N-tile; weight tile
// (32 rows x 32 k) staged in double-buffered, bank-padded LDS by wave 0
// (async if available); A fragments double-buffered in registers.
// acc[mt][nt] covers rows m0+mt*16, cols n0+nt*16.
// ---------------------------------------------------------------------------
struct GemmCore {
  v8f acc[4][2];
  int m0, n0, lane, lr, lk, rrow;

  __device__ inline void run(const half_t* __restrict__ ah,
                             const half_t* __restrict__ wh,
                             half_t (*bsh)[32 * LDSP], int wv) {
    lane = threadIdx.x & 31;
    lr = lane & 15;
    lk = (lane >> 4) << 4;
    rrow = (lane >> 4) << 3;
#pragma unroll
    for (int mt = 0; mt < 4; ++mt)
#pragma unroll
      for (int nt = 0; nt < 2; ++nt) acc[mt][nt] = {};

    const half_t* apx = ah + (size_t)(m0 + lr) * Cx + lk;

    // stage one 32x32 weight tile (2 KB) into bsh[buf] (padded rows)
    auto stage = [&](int k, int buf) {
#pragma unroll
      for (int i = 0; i < 4; ++i) {
        const int chunk = i * 32 + lane;       // 0..127 16-byte chunks
        const int row   = chunk >> 2;          // 0..31 weight row (n0+row)
        const int cc    = (chunk & 3) << 3;    // half offset 0,8,16,24
        const half_t* src = wh + (size_t)(n0 + row) * Cx + k + cc;
#if USE_ASYNC_LDS
        __builtin_amdgcn_global_load_async_to_lds_b128(
            (AS1 v4i_g*)src, (AS3 v4i_g*)&bsh[buf][row * LDSP + cc], 0, 0);
#else
        *(uint4*)&bsh[buf][row * LDSP + cc] = *(const uint4*)src;
#endif
      }
    };

    v16h a[2][4];
#pragma unroll
    for (int mt = 0; mt < 4; ++mt)
      a[0][mt] = *(const v16h*)(apx + (size_t)mt * 16 * Cx);
    if (wv == 0) stage(0, 0);

    for (int ks = 0; ks < Cx / 32; ++ks) {
      const int cur = ks & 1;
      const int k = ks * 32;
#if USE_ASYNC_LDS
      if (wv == 0) __builtin_amdgcn_s_wait_asynccnt(0);
#endif
      __syncthreads();                          // bsh[cur] ready
      if (k + 32 < Cx) {
        if (wv == 0) stage(k + 32, cur ^ 1);    // prefetch next weight tile
        __builtin_prefetch(apx + k + 64, 0, 1); // stream hint for A
#pragma unroll
        for (int mt = 0; mt < 4; ++mt)          // next A frags in flight
          a[cur ^ 1][mt] = *(const v16h*)(apx + k + 32 + (size_t)mt * 16 * Cx);
      }
      const v16h b0 = lds_frag(&bsh[cur][lr * LDSP + lk]);
      const v16h b1 = lds_frag(&bsh[cur][(16 + lr) * LDSP + lk]);
#pragma unroll
      for (int mt = 0; mt < 4; ++mt) {
        acc[mt][0] = WMMA_F16(a[cur][mt], b0, acc[mt][0]);
        acc[mt][1] = WMMA_F16(a[cur][mt], b1, acc[mt][1]);
      }
      __syncthreads();                          // bsh[cur] consumed
    }
  }
};

// ---------------------------------------------------------------------------
// QKV GEMM: out[m][n] = x[m,:]·w_qkv[n,:] + b_qkv[n].
// Q scaled + stored [B][H][N][D]; K stored [B][H][N][D]; V stored transposed
// [B][H][D][N] so attention P·V B-fragments are contiguous loads.
// ---------------------------------------------------------------------------
__global__ __launch_bounds__(256) void qkv_gemm(
    const half_t* __restrict__ xh, const half_t* __restrict__ wh,
    const float* __restrict__ bias,
    half_t* __restrict__ qh, half_t* __restrict__ kh, half_t* __restrict__ vt) {
  __shared__ half_t bsh[2][32 * LDSP];
  const int wv = threadIdx.x >> 5;
  GemmCore g;
  g.n0 = (blockIdx.x % (QKV_N / 32)) * 32;
  g.m0 = (blockIdx.x / (QKV_N / 32)) * 512 + wv * 64;
  g.run(xh, wh, bsh, wv);

  // 64-row wave block lies inside one batch (320 = 5*64): hoist batch/token
  const int bb   = g.m0 / Nx;
  const int tok0 = g.m0 % Nx;
#pragma unroll
  for (int nt = 0; nt < 2; ++nt) {
    const int n     = g.n0 + nt * 16;
    const int which = n / Cx;                 // uniform per wave: 0=Q 1=K 2=V
    const int f     = n % Cx;
    const int h     = f >> 6;
    const int d     = (f & 63) + g.lr;
    const int bh    = bb * Hx + h;
    const float bv  = bias[n + g.lr];
    if (which == 2) {
      // V^T: per lane, 8 consecutive tokens at fixed d -> paired stores
      half_t* dst = vt + ((size_t)bh * Dx + d) * Nx + tok0 + g.rrow;
#pragma unroll
      for (int mt = 0; mt < 4; ++mt)
#pragma unroll
        for (int r = 0; r < 8; r += 2) {
          v2h p;
          p.x = (half_t)(g.acc[mt][nt][r] + bv);
          p.y = (half_t)(g.acc[mt][nt][r + 1] + bv);
          *(v2h*)&dst[mt * 16 + r] = p;
        }
    } else {
      half_t* dst = (which == 0 ? qh : kh) +
                    ((size_t)(bh * Nx + tok0 + g.rrow)) * Dx + d;
      const float sc = (which == 0) ? Q_SCALE : 1.0f;
#pragma unroll
      for (int mt = 0; mt < 4; ++mt)
#pragma unroll
        for (int r = 0; r < 8; ++r)
          dst[(size_t)(mt * 16 + r) * Dx] = (half_t)((g.acc[mt][nt][r] + bv) * sc);
    }
  }
}

// ---------------------------------------------------------------------------
// Flash attention: one wave per (b*H+h, 16-query tile).
// qt in [0,4): temporal (keys 0..63). qt in [4,20): spatial (keys 0..319).
// ---------------------------------------------------------------------------
__global__ __launch_bounds__(32) void flash_attn(
    const half_t* __restrict__ qh, const half_t* __restrict__ kh,
    const half_t* __restrict__ vt, half_t* __restrict__ xatt) {
  __shared__ half_t p_lds[16 * LDSP];        // 16 q-rows x 32 keys, padded

  const int lane = threadIdx.x;
  const int bh = blockIdx.x;                 // 0..767
  const int b  = bh / Hx;
  const int h  = bh % Hx;
  const int qt = blockIdx.y;                 // 0..19
  const int t0 = qt * 16;
  const int Nk = (qt < 4) ? 64 : Nx;
  const int lr = lane & 15;
  const int lk = (lane >> 4) << 4;
  const int rrow = (lane >> 4) << 3;

  const half_t* Q  = qh + (size_t)bh * Nx * Dx;
  const half_t* K  = kh + (size_t)bh * Nx * Dx;
  const half_t* Vt = vt + (size_t)bh * Dx * Nx;

  const v16h aq0 = *(const v16h*)(Q + (t0 + lr) * Dx + 0  + lk);
  const v16h aq1 = *(const v16h*)(Q + (t0 + lr) * Dx + 32 + lk);

  v8f o[4];
#pragma unroll
  for (int n = 0; n < 4; ++n) o[n] = {};
  float m[8], l[8];
#pragma unroll
  for (int r = 0; r < 8; ++r) { m[r] = -1e30f; l[r] = 0.0f; }

  for (int k0 = 0; k0 < Nk; k0 += 32) {
    // ---- S = Q * K^T over 32 keys (two 16-key column tiles) ----
    v8f c0 = {}, c1 = {};
    {
      const half_t* Kp = K + (size_t)(k0 + lr) * Dx + lk;
      v16h b0 = *(const v16h*)(Kp);
      v16h b1 = *(const v16h*)(Kp + 16 * Dx);
      c0 = WMMA_F16(aq0, b0, c0);
      c1 = WMMA_F16(aq0, b1, c1);
      b0 = *(const v16h*)(Kp + 32);
      b1 = *(const v16h*)(Kp + 16 * Dx + 32);
      c0 = WMMA_F16(aq1, b0, c0);
      c1 = WMMA_F16(aq1, b1, c1);
    }

    // ---- online softmax (exp2 domain); columns live across lanes 0..15 ----
    float corr[8];
#pragma unroll
    for (int r = 0; r < 8; ++r) {
      float t = fmaxf(c0[r], c1[r]);
      t = fmaxf(t, __shfl_xor(t, 1, 32));
      t = fmaxf(t, __shfl_xor(t, 2, 32));
      t = fmaxf(t, __shfl_xor(t, 4, 32));
      t = fmaxf(t, __shfl_xor(t, 8, 32));
      const float mn = fmaxf(m[r], t);
      corr[r] = __builtin_amdgcn_exp2f(m[r] - mn);
      m[r] = mn;
      const float p0 = __builtin_amdgcn_exp2f(c0[r] - mn);
      const float p1 = __builtin_amdgcn_exp2f(c1[r] - mn);
      float s = p0 + p1;
      s += __shfl_xor(s, 1, 32);
      s += __shfl_xor(s, 2, 32);
      s += __shfl_xor(s, 4, 32);
      s += __shfl_xor(s, 8, 32);
      l[r] = l[r] * corr[r] + s;
      const int row = r + rrow;
      p_lds[row * LDSP + lr]      = (half_t)p0;
      p_lds[row * LDSP + 16 + lr] = (half_t)p1;
    }
    __syncthreads();

    // ---- reload P in A-fragment layout, rescale O, accumulate P*V ----
    const v16h ap = lds_frag(&p_lds[lr * LDSP + lk]);
#pragma unroll
    for (int n = 0; n < 4; ++n) {
#pragma unroll
      for (int r = 0; r < 8; ++r) o[n][r] *= corr[r];
      const v16h bv = *(const v16h*)(Vt + (size_t)(n * 16 + lr) * Nx + k0 + lk);
      o[n] = WMMA_F16(ap, bv, o[n]);
    }
    __syncthreads();
  }

  // ---- normalize and store to x_att [B][N][C] as f16 ----
  float invl[8];
#pragma unroll
  for (int r = 0; r < 8; ++r) invl[r] = 1.0f / l[r];
  half_t* dst = xatt + ((size_t)(b * Nx + t0 + rrow)) * Cx + h * Dx + lr;
#pragma unroll
  for (int n = 0; n < 4; ++n)
#pragma unroll
    for (int r = 0; r < 8; ++r)
      dst[(size_t)r * Cx + n * 16] = (half_t)(o[n][r] * invl[r]);
}

// ---------------------------------------------------------------------------
// Output projection: out[m][n] = x_att[m,:]·w_proj[n,:] + b_proj[n], f32 out.
// ---------------------------------------------------------------------------
__global__ __launch_bounds__(256) void proj_gemm(
    const half_t* __restrict__ xh, const half_t* __restrict__ wh,
    const float* __restrict__ bias, float* __restrict__ out) {
  __shared__ half_t bsh[2][32 * LDSP];
  const int wv = threadIdx.x >> 5;
  GemmCore g;
  g.n0 = (blockIdx.x % (Cx / 32)) * 32;
  g.m0 = (blockIdx.x / (Cx / 32)) * 512 + wv * 64;
  g.run(xh, wh, bsh, wv);

#pragma unroll
  for (int nt = 0; nt < 2; ++nt) {
    const float bv = bias[g.n0 + nt * 16 + g.lr];
    float* dst = out + (size_t)(g.m0 + g.rrow) * Cx + g.n0 + nt * 16 + g.lr;
#pragma unroll
    for (int mt = 0; mt < 4; ++mt)
#pragma unroll
      for (int r = 0; r < 8; ++r)
        dst[(size_t)(mt * 16 + r) * Cx] = g.acc[mt][nt][r] + bv;
  }
}

// ---------------------------------------------------------------------------
// Host launcher
// ---------------------------------------------------------------------------
extern "C" void kernel_launch(void* const* d_in, const int* in_sizes, int n_in,
                              void* d_out, int out_size, void* d_ws, size_t ws_size,
                              hipStream_t stream) {
  const float* x      = (const float*)d_in[0];   // [64,320,768]
  const float* w_qkv  = (const float*)d_in[1];   // [2304,768]
  const float* b_qkv  = (const float*)d_in[2];   // [2304]
  const float* w_proj = (const float*)d_in[3];   // [768,768]
  const float* b_proj = (const float*)d_in[4];   // [768]
  float* out = (float*)d_out;                    // [64,320,768]

  char* ws = (char*)d_ws;
  const size_t SZ_XH   = (size_t)M_TOK * Cx * sizeof(half_t);
  const size_t SZ_WQKV = (size_t)QKV_N * Cx * sizeof(half_t);
  const size_t SZ_WPRJ = (size_t)Cx * Cx * sizeof(half_t);
  const size_t SZ_QKV1 = (size_t)Bx * Hx * Nx * Dx * sizeof(half_t);

  size_t off = 0;
  half_t* x_h    = (half_t*)(ws + off); off += SZ_XH;
  half_t* wqkv_h = (half_t*)(ws + off); off += SZ_WQKV;
  half_t* wprj_h = (half_t*)(ws + off); off += SZ_WPRJ;
  half_t* q_h    = (half_t*)(ws + off); off += SZ_QKV1;
  half_t* k_h    = (half_t*)(ws + off); off += SZ_QKV1;
  half_t* vt_h   = (half_t*)(ws + off); off += SZ_QKV1;
  half_t* x_att  = (half_t*)(ws + off); off += SZ_XH;

  // 1) f32 -> f16 conversions (4-wide)
  cvt_f32_to_f16<<<2048, 256, 0, stream>>>(x,      x_h,    M_TOK * Cx / 4);
  cvt_f32_to_f16<<<512,  256, 0, stream>>>(w_qkv,  wqkv_h, QKV_N * Cx / 4);
  cvt_f32_to_f16<<<256,  256, 0, stream>>>(w_proj, wprj_h, Cx * Cx / 4);

  // 2) QKV projection: blocks of 8 waves, 512(M)x32(N) per block
  qkv_gemm<<<(M_TOK / 512) * (QKV_N / 32), 256, 0, stream>>>(
      x_h, wqkv_h, b_qkv, q_h, k_h, vt_h);

  // 3) Flash attention: (b*h) x 20 query tiles, one wave each
  flash_attn<<<dim3(Bx * Hx, 20), 32, 0, stream>>>(q_h, k_h, vt_h, x_att);

  // 4) Output projection
  proj_gemm<<<(M_TOK / 512) * (Cx / 32), 256, 0, stream>>>(
      x_att, wprj_h, b_proj, out);
}
